// VectorQuantizer_10746008174849
// MI455X (gfx1250) — matrix operand order repaired and checked by reading, not compiled, not run on hardware
//
#include <hip/hip_runtime.h>
#include <hip/hip_bf16.h>

typedef __attribute__((ext_vector_type(16))) __bf16   v16bf;
typedef __attribute__((ext_vector_type(8)))  float    v8f;
typedef __attribute__((ext_vector_type(8)))  uint32_t v8u;

#define THREADS       128          // 4 waves of 32
#define ROWS_PER_BLK  64           // pixels per workgroup
#define C_DIM         256
#define K_CODES       1024
#define KTILES        (K_CODES / 16)
#define LDSX_STRIDE   132          // dwords per row (264 bf16, padded from 256) -> stride%64==4, conflict-free
#define LDSB_STRIDE   132

// ---------------------------------------------------------------------------
// Kernel 0: e_sq (fp32), codebook fp32->bf16 into ws, zero the loss slot.
// ---------------------------------------------------------------------------
__global__ void vq_prep(const float* __restrict__ cb,
                        float* __restrict__ esq,
                        __bf16* __restrict__ cbbf,
                        float* __restrict__ loss_slot) {
    int code = blockIdx.x * blockDim.x + threadIdx.x;
    if (code == 0) *loss_slot = 0.0f;
    if (code < K_CODES) {
        const float* row = cb + (size_t)code * C_DIM;
        __bf16* orow = cbbf + (size_t)code * C_DIM;
        float s = 0.0f;
        for (int c = 0; c < C_DIM; ++c) {
            float v = row[c];
            s = fmaf(v, v, s);
            orow[c] = (__bf16)v;
        }
        esq[code] = s;
    }
}

// ---------------------------------------------------------------------------
// Kernel 1: distances via bf16 WMMA + fused argmin + gather/STE/loss.
// ---------------------------------------------------------------------------
__global__ __launch_bounds__(THREADS) void vq_main(
    const float*    __restrict__ x,      // [32,256,64,64] NCHW
    const float*    __restrict__ cb,     // [1024,256] fp32
    const float*    __restrict__ esq,    // [1024]
    const uint32_t* __restrict__ cbbf,   // [1024,256] bf16, viewed as u32 pairs
    float*          __restrict__ out,    // [32,256,64,64]
    float*          __restrict__ loss_slot) {

    __shared__ uint32_t sh_x[ROWS_PER_BLK * LDSX_STRIDE]; // 64 rows x 264 bf16 (padded)
    __shared__ uint32_t sh_b[2 * 16 * LDSB_STRIDE];       // double-buffered 16-code tiles
    __shared__ float    sh_esq[K_CODES];
    __shared__ int      sh_idx[ROWS_PER_BLK];
    __shared__ float    sh_red[THREADS / 32];

    const int tid  = threadIdx.x;
    const int lane = tid & 31;
    const int wave = tid >> 5;

    const int n0   = blockIdx.x * ROWS_PER_BLK;   // 64 consecutive pixels, same image
    const int b    = n0 >> 12;                    // 64*64 = 4096 pixels/image
    const int hw0  = n0 & 4095;
    const float* xb = x + (size_t)b * C_DIM * 4096 + hw0;

    // ---- stage x tile fp32 -> bf16 in LDS (coalesced over hw) ----
    __bf16* shx_h = reinterpret_cast<__bf16*>(sh_x);
    for (int e = tid; e < ROWS_PER_BLK * C_DIM; e += THREADS) {
        int r = e & 63;                 // pixel (contiguous per lane -> 256B lines)
        int c = e >> 6;                 // channel (stride 4096 floats)
        float v = xb[(size_t)c * 4096 + r];
        shx_h[r * (LDSX_STRIDE * 2) + c] = (__bf16)v;
    }
    for (int e = tid; e < K_CODES; e += THREADS) sh_esq[e] = esq[e];

    // prefetch code tile 0 into registers (thread owns dword-column `tid`)
    uint32_t pf[16];
#pragma unroll
    for (int i = 0; i < 16; ++i)
        pf[i] = cbbf[(size_t)i * 128 + tid];

    __syncthreads();

    // ---- build resident A fragments: 16 rows x 256 channels per wave ----
    // ISA 16-bit A 16x32 layout: lanes 0-15 -> M=lane, K pairs {0..7,16..23};
    // lanes 16-31 -> M=lane-16, K pairs {8..15,24..31}.
    const int half = lane >> 4;
    const int mrow = lane & 15;
    const int arow = wave * 16 + mrow;
    v16bf afrag[8];
#pragma unroll
    for (int kk = 0; kk < 8; ++kk) {
        v8u t;
#pragma unroll
        for (int v = 0; v < 8; ++v) {
            int k = kk * 32 + ((v < 4) ? 2 * v : 16 + 2 * (v - 4)) + 8 * half;
            t[v] = sh_x[arow * LDSX_STRIDE + (k >> 1)];
        }
        afrag[kk] = __builtin_bit_cast(v16bf, t);
    }

    float minv[8];
    int   mini[8];
#pragma unroll
    for (int r = 0; r < 8; ++r) { minv[r] = 3.4e38f; mini[r] = 0; }

    const int ncol = lane & 15;   // D-layout: column N = lane%16

    // ---- sweep the 64 code tiles (double-buffered staging) ----
    for (int kt = 0; kt < KTILES; ++kt) {
        uint32_t* sb = sh_b + (kt & 1) * (16 * LDSB_STRIDE);

        // commit prefetched tile kt to LDS. The implicit s_wait_loadcnt here
        // drains loads issued at kt-1, which had the whole compute(kt-1)
        // phase to complete. Barrier(kt-1) separated this overwrite from all
        // waves' compute on this buffer at kt-2.
#pragma unroll
        for (int i = 0; i < 16; ++i)
            sb[i * LDSB_STRIDE + tid] = pf[i];

        __syncthreads();

        // issue loads for tile kt+1 AFTER the barrier so the conservative
        // pre-barrier s_wait_loadcnt does not drain them; they stay in
        // flight across the whole compute(kt) phase below.
        if (kt + 1 < KTILES) {
#pragma unroll
            for (int i = 0; i < 16; ++i)
                pf[i] = cbbf[(size_t)((kt + 1) * 16 + i) * 128 + tid];
        }

        v8f acc = {0.f, 0.f, 0.f, 0.f, 0.f, 0.f, 0.f, 0.f};
#pragma unroll
        for (int kk = 0; kk < 8; ++kk) {
            // B 32x16 layout: lanes 0-15 -> N=lane, K {0..7,16..23}; lanes 16-31 -> K {8..15,24..31}
            v8u t;
#pragma unroll
            for (int v = 0; v < 8; ++v) {
                int k = kk * 32 + ((v < 4) ? 2 * v : 16 + 2 * (v - 4)) + 8 * half;
                t[v] = sb[ncol * LDSB_STRIDE + (k >> 1)];
            }
            v16bf bfrag = __builtin_bit_cast(v16bf, t);
            acc = __builtin_amdgcn_wmma_f32_16x16x32_bf16(
                false, afrag[kk], false, bfrag, (short)0, acc, false, false);
        }

        int   code = kt * 16 + ncol;
        float es   = sh_esq[code];
#pragma unroll
        for (int r = 0; r < 8; ++r) {
            float d = es - 2.0f * acc[r];           // x_sq omitted: constant per row
            if (d < minv[r]) { minv[r] = d; mini[r] = code; }
        }
    }

    // ---- argmin across the 16 lanes holding one row (within each half) ----
#pragma unroll
    for (int r = 0; r < 8; ++r) {
#pragma unroll
        for (int off = 8; off >= 1; off >>= 1) {
            float ov = __shfl_xor(minv[r], off, 32);
            int   oi = __shfl_xor(mini[r], off, 32);
            if (ov < minv[r] || (ov == minv[r] && oi < mini[r])) {
                minv[r] = ov; mini[r] = oi;
            }
        }
    }
    if ((lane & 15) == 0) {
#pragma unroll
        for (int r = 0; r < 8; ++r)
            sh_idx[wave * 16 + half * 8 + r] = mini[r];  // VGPR r -> rows r, r+8
    }
    __syncthreads();

    // ---- gather + straight-through output (NCHW) + fused loss ----
    float lsum = 0.0f;
    float* outb = out + (size_t)b * C_DIM * 4096 + hw0;
    for (int e = tid; e < ROWS_PER_BLK * C_DIM; e += THREADS) {
        int r = e & 63;
        int c = e >> 6;
        size_t ga = (size_t)c * 4096 + r;
        float q  = cb[(size_t)sh_idx[r] * C_DIM + c];
        float xv = xb[ga];
        float dq = q - xv;
        outb[ga] = xv + dq;                 // STE: x + sg(q - x)
        lsum = fmaf(dq, dq, lsum);
    }
#pragma unroll
    for (int off = 16; off >= 1; off >>= 1) lsum += __shfl_xor(lsum, off, 32);
    if (lane == 0) sh_red[wave] = lsum;
    __syncthreads();
    if (tid == 0) {
        float s = sh_red[0] + sh_red[1] + sh_red[2] + sh_red[3];
        // loss = (1 + 0.25) * mean(dq^2) over 32*256*64*64 elements
        atomicAdd(loss_slot, s * (1.25f / 33554432.0f));
    }
}

// ---------------------------------------------------------------------------
extern "C" void kernel_launch(void* const* d_in, const int* in_sizes, int n_in,
                              void* d_out, int out_size, void* d_ws, size_t ws_size,
                              hipStream_t stream) {
    const float* x  = (const float*)d_in[0];   // [32,256,64,64]
    const float* cb = (const float*)d_in[1];   // [1024,256]
    float* out = (float*)d_out;
    float* loss_slot = out + (out_size - 1);   // tuple: [quantized_out..., loss]

    // ws layout: [0,4KB) e_sq fp32 ; [4KB, 4KB+512KB) codebook bf16
    float*  esq  = (float*)d_ws;
    __bf16* cbbf = (__bf16*)((char*)d_ws + 4096);

    vq_prep<<<8, 128, 0, stream>>>(cb, esq, cbbf, loss_slot);
    vq_main<<<131072 / ROWS_PER_BLK, THREADS, 0, stream>>>(
        x, cb, esq, (const uint32_t*)cbbf, out, loss_slot);
}
